// gmlp_65541200937048
// MI455X (gfx1250) — matrix operand order repaired and checked by reading, not compiled, not run on hardware
//
#include <hip/hip_runtime.h>
#include <hip/hip_bf16.h>

// Problem sizes
#define BB 16
#define SS 1024
#define HH 1024
#define FF 2048            // 2*S
#define TT 1024            // channel dim of u/v
#define M1 (BB * SS)       // 16384 rows

typedef __attribute__((ext_vector_type(16))) __bf16 v16bf;
typedef __attribute__((ext_vector_type(8)))  __bf16 v8bf;
typedef __attribute__((ext_vector_type(8)))  float  v8f;
typedef __attribute__((ext_vector_type(4)))  float  v4f;

union V16U { v16bf v; v8bf h[2]; };

// ---------------------------------------------------------------------------
// Core: one wave computes a 64x64 f32 accumulator patch of C = A @ B
//   A  : row-major bf16 [M][K], lda = K stride in elements (compile-time const
//        at all call sites so offsets fold into load immediates)
//   BT : row-major bf16 [N][K] (i.e. B transposed), ldb = K stride
// Fragment layouts used here (contiguous k, same order in A and B):
//   A 16x32: lane<16 holds K {0..15} of row M=lane; lane>=16 holds
//            K {16..31} of row M=lane-16  -> two 16B loads
//   B 32x16: lane<16 holds K {0..15} of col N=lane; lane>=16 holds K{16..31}
//            -> one 32B load from BT
// Per 32-K step: 4 A frags + 4 B frags (16 x 16B loads) feed 16 WMMAs.
// ---------------------------------------------------------------------------
__device__ __forceinline__ void wmma_patch_64x64(
    const __bf16* __restrict__ A, int lda,
    const __bf16* __restrict__ BT, int ldb,
    int K, int rowBase, int colBase, int lane,
    v8f acc[4][4])
{
    const int lhalf = lane >> 4;     // 0 or 1
    const int l16   = lane & 15;

    const __bf16* ap = A  + (size_t)(rowBase + l16) * lda + (lhalf << 4);
    const __bf16* bp = BT + (size_t)(colBase + l16) * ldb + (lhalf << 4);

    const v8f zero = {};
    #pragma unroll
    for (int i = 0; i < 4; ++i)
        #pragma unroll
        for (int j = 0; j < 4; ++j)
            acc[i][j] = zero;

    for (int k = 0; k < K; k += 32) {
        V16U a[4], b[4];
        #pragma unroll
        for (int i = 0; i < 4; ++i) {
            const __bf16* p = ap + (size_t)(16 * i) * lda + k;
            a[i].h[0] = *(const v8bf*)(p);
            a[i].h[1] = *(const v8bf*)(p + 8);
        }
        #pragma unroll
        for (int j = 0; j < 4; ++j) {
            const __bf16* p = bp + (size_t)(16 * j) * ldb + k;
            b[j].v = *(const v16bf*)(p);
        }

        // prefetch next K-step (speculative, OOB silently dropped on gfx1250)
        __builtin_prefetch(ap + k + 64, 0, 1);
        __builtin_prefetch(bp + k + 64, 0, 1);

        #pragma unroll
        for (int i = 0; i < 4; ++i)
            #pragma unroll
            for (int j = 0; j < 4; ++j)
                acc[i][j] = __builtin_amdgcn_wmma_f32_16x16x32_bf16(
                    false, a[i].v, false, b[j].v, (short)0, acc[i][j],
                    false, false);
    }
}

// ---------------------------------------------------------------------------
// Transpose + f32 -> bf16 convert:  out[c][r] = (bf16) in[r][c]
// Dims must be multiples of 32 (all are here). grid.z = batch.
// ---------------------------------------------------------------------------
__global__ __launch_bounds__(256) void transpose_cvt_kernel(
    const float* __restrict__ in, __bf16* __restrict__ out,
    int R, int C, long inBatchStride, long outBatchStride)
{
    __shared__ float tile[32][33];
    const float* inb = in  + (size_t)blockIdx.z * inBatchStride;
    __bf16*     outb = out + (size_t)blockIdx.z * outBatchStride;
    int c0 = blockIdx.x * 32, r0 = blockIdx.y * 32;
    int tx = threadIdx.x & 31;
    int ty = threadIdx.x >> 5;           // 0..7
    for (int k = 0; k < 32; k += 8)
        tile[ty + k][tx] = inb[(size_t)(r0 + ty + k) * C + (c0 + tx)];
    __syncthreads();
    for (int k = 0; k < 32; k += 8)
        outb[(size_t)(c0 + ty + k) * R + (r0 + tx)] = (__bf16)tile[tx][ty + k];
}

// ---------------------------------------------------------------------------
// LN1: per-row layernorm of x over H, output bf16 (A operand of GEMM1)
// one block per row, 256 threads, 4 elems/thread
// ---------------------------------------------------------------------------
__global__ __launch_bounds__(256) void ln1_kernel(
    const float* __restrict__ x, const float* __restrict__ ng,
    const float* __restrict__ nb, __bf16* __restrict__ out)
{
    __shared__ float s_sum[256], s_sq[256];
    const int row = blockIdx.x;
    const int tid = threadIdx.x;
    const float* xr = x + (size_t)row * HH;
    v4f xv = ((const v4f*)xr)[tid];
    float sum = xv[0] + xv[1] + xv[2] + xv[3];
    float sq  = xv[0]*xv[0] + xv[1]*xv[1] + xv[2]*xv[2] + xv[3]*xv[3];
    s_sum[tid] = sum; s_sq[tid] = sq;
    __syncthreads();
    for (int off = 128; off > 0; off >>= 1) {
        if (tid < off) { s_sum[tid] += s_sum[tid+off]; s_sq[tid] += s_sq[tid+off]; }
        __syncthreads();
    }
    float mean = s_sum[0] * (1.0f / HH);
    float var  = s_sq[0] * (1.0f / HH) - mean * mean;
    float rstd = rsqrtf(var + 1e-5f);
    int base = tid * 4;
    __bf16* orow = out + (size_t)row * HH;
    #pragma unroll
    for (int i = 0; i < 4; ++i) {
        int hcol = base + i;
        float val = (xv[i] - mean) * rstd * ng[hcol] + nb[hcol];
        orow[hcol] = (__bf16)val;
    }
}

// ---------------------------------------------------------------------------
// LN2 (in place on v, f32): vrow = layernorm(vrow; sng,snb) * weight[row]
// ---------------------------------------------------------------------------
__global__ __launch_bounds__(256) void ln2_kernel(
    float* __restrict__ v, const float* __restrict__ sng,
    const float* __restrict__ snb, const float* __restrict__ weight)
{
    __shared__ float s_sum[256], s_sq[256];
    const int row = blockIdx.x;           // b*S + s
    const int tid = threadIdx.x;
    float* vr = v + (size_t)row * TT;
    v4f xv = ((const v4f*)vr)[tid];
    float sum = xv[0] + xv[1] + xv[2] + xv[3];
    float sq  = xv[0]*xv[0] + xv[1]*xv[1] + xv[2]*xv[2] + xv[3]*xv[3];
    s_sum[tid] = sum; s_sq[tid] = sq;
    __syncthreads();
    for (int off = 128; off > 0; off >>= 1) {
        if (tid < off) { s_sum[tid] += s_sum[tid+off]; s_sq[tid] += s_sq[tid+off]; }
        __syncthreads();
    }
    float mean = s_sum[0] * (1.0f / TT);
    float var  = s_sq[0] * (1.0f / TT) - mean * mean;
    float rstd = rsqrtf(var + 1e-5f);
    float wg   = weight[row];
    int base = tid * 4;
    v4f o;
    #pragma unroll
    for (int i = 0; i < 4; ++i) {
        int t = base + i;
        o[i] = ((xv[i] - mean) * rstd * sng[t] + snb[t]) * wg;
    }
    ((v4f*)vr)[tid] = o;
}

// ---------------------------------------------------------------------------
// GEMM1: C = xln(16384x1024) @ pw(1024x2048) + pb, SiLU, split to u/v
// block tile 128(M) x 256(N); wave tile 64x64; waves 2x4
// ---------------------------------------------------------------------------
__global__ __launch_bounds__(256) void gemm1_silu_kernel(
    const __bf16* __restrict__ xln,    // [M1][HH]
    const __bf16* __restrict__ pwT,    // [FF][HH]
    const float*  __restrict__ pb,     // [FF]
    __bf16* __restrict__ u,            // [M1][TT]
    float*  __restrict__ v)            // [M1][TT]
{
    const int lane = threadIdx.x & 31;
    const int wave = threadIdx.x >> 5;
    const int rowBase = blockIdx.y * 128 + (wave >> 2) * 64;
    const int colBase = blockIdx.x * 256 + (wave & 3) * 64;
    v8f acc[4][4];
    wmma_patch_64x64(xln, HH, pwT, HH, HH, rowBase, colBase, lane, acc);

    const int n0   = colBase + (lane & 15);
    const int rOff = (lane >> 4) * 8;
    #pragma unroll
    for (int tm = 0; tm < 4; ++tm) {
        #pragma unroll
        for (int tn = 0; tn < 4; ++tn) {
            const int n = n0 + tn * 16;
            const float bias = pb[n];
            #pragma unroll
            for (int r = 0; r < 8; ++r) {
                const int m = rowBase + tm * 16 + rOff + r;
                float val = acc[tm][tn][r] + bias;
                val = val / (1.0f + __expf(-val));      // SiLU
                if (n < TT) u[(size_t)m * TT + n] = (__bf16)val;
                else        v[(size_t)m * TT + (n - TT)] = val;
            }
        }
    }
}

// ---------------------------------------------------------------------------
// GEMM2 (per batch): gated = sgu_w^T (j,s) @ m (s,t);
// v2 = gated*wg[b,j] + sgu_b[j]; proj = u * v2 (bf16 out)
// ---------------------------------------------------------------------------
__global__ __launch_bounds__(256) void gemm2_gate_kernel(
    const __bf16* __restrict__ sguT,   // [SS][SS]  (j, s)
    const __bf16* __restrict__ mT,     // [BB][TT][SS]
    const __bf16* __restrict__ u,      // [M1][TT]
    const float*  __restrict__ weight, // [BB*SS]
    const float*  __restrict__ sgu_b,  // [SS]
    __bf16* __restrict__ proj)         // [M1][TT]
{
    const int b = blockIdx.z;
    const __bf16* mTb = mT + (size_t)b * TT * SS;
    const int lane = threadIdx.x & 31;
    const int wave = threadIdx.x >> 5;
    const int rowBase = blockIdx.y * 128 + (wave >> 2) * 64;   // j
    const int colBase = blockIdx.x * 256 + (wave & 3) * 64;    // t
    v8f acc[4][4];
    wmma_patch_64x64(sguT, SS, mTb, SS, SS, rowBase, colBase, lane, acc);

    const int n0   = colBase + (lane & 15);
    const int rOff = (lane >> 4) * 8;
    #pragma unroll
    for (int tm = 0; tm < 4; ++tm) {
        #pragma unroll
        for (int tn = 0; tn < 4; ++tn) {
            const int t = n0 + tn * 16;
            #pragma unroll
            for (int r = 0; r < 8; ++r) {
                const int j = rowBase + tm * 16 + rOff + r;
                const float wg = weight[b * SS + j];
                float v2 = acc[tm][tn][r] * wg + sgu_b[j];
                const size_t idx = ((size_t)b * SS + j) * TT + t;
                float uf = (float)u[idx];
                proj[idx] = (__bf16)(uf * v2);
            }
        }
    }
}

// ---------------------------------------------------------------------------
// GEMM3: out = proj(16384x1024) @ cw(1024x1024) + cb   (f32 out)
// ---------------------------------------------------------------------------
__global__ __launch_bounds__(256) void gemm3_out_kernel(
    const __bf16* __restrict__ proj,   // [M1][TT]
    const __bf16* __restrict__ cwT,    // [HH][TT]
    const float*  __restrict__ cb,     // [HH]
    float* __restrict__ out)           // [M1][HH]
{
    const int lane = threadIdx.x & 31;
    const int wave = threadIdx.x >> 5;
    const int rowBase = blockIdx.y * 128 + (wave >> 2) * 64;
    const int colBase = blockIdx.x * 256 + (wave & 3) * 64;
    v8f acc[4][4];
    wmma_patch_64x64(proj, TT, cwT, TT, TT, rowBase, colBase, lane, acc);

    const int n0   = colBase + (lane & 15);
    const int rOff = (lane >> 4) * 8;
    #pragma unroll
    for (int tm = 0; tm < 4; ++tm) {
        #pragma unroll
        for (int tn = 0; tn < 4; ++tn) {
            const int n = n0 + tn * 16;
            const float bias = cb[n];
            #pragma unroll
            for (int r = 0; r < 8; ++r) {
                const int m = rowBase + tm * 16 + rOff + r;
                out[(size_t)m * HH + n] = acc[tm][tn][r] + bias;
            }
        }
    }
}

// ---------------------------------------------------------------------------
extern "C" void kernel_launch(void* const* d_in, const int* in_sizes, int n_in,
                              void* d_out, int out_size, void* d_ws, size_t ws_size,
                              hipStream_t stream) {
    const float* x      = (const float*)d_in[0];   // (B,S,H)
    const float* weight = (const float*)d_in[1];   // (B,S)
    const float* ng     = (const float*)d_in[2];   // (H,)
    const float* nb     = (const float*)d_in[3];   // (H,)
    const float* pw     = (const float*)d_in[4];   // (H,2S)
    const float* pb     = (const float*)d_in[5];   // (2S,)
    const float* sng    = (const float*)d_in[6];   // (S,)
    const float* snb    = (const float*)d_in[7];   // (S,)
    const float* sgu_w  = (const float*)d_in[8];   // (S,S)
    const float* sgu_b  = (const float*)d_in[9];   // (S,)
    const float* cw     = (const float*)d_in[10];  // (S,H)
    const float* cb     = (const float*)d_in[11];  // (H,)
    float* out = (float*)d_out;

    // Workspace layout
    size_t off = 0;
    auto take = [&](size_t bytes) {
        void* p = (char*)d_ws + off;
        off += (bytes + 255) & ~(size_t)255;
        return p;
    };
    __bf16* xln   = (__bf16*)take((size_t)M1 * HH * 2);   // 32 MB
    __bf16* pwT   = (__bf16*)take((size_t)FF * HH * 2);   //  4 MB
    __bf16* cwT   = (__bf16*)take((size_t)HH * TT * 2);   //  2 MB
    __bf16* sguT  = (__bf16*)take((size_t)SS * SS * 2);   //  2 MB
    __bf16* ubuf  = (__bf16*)take((size_t)M1 * TT * 2);   // 32 MB
    float*  vbuf  = (float*) take((size_t)M1 * TT * 4);   // 64 MB (reused as m)
    __bf16* mT    = (__bf16*)take((size_t)BB * TT * SS * 2); // 32 MB
    __bf16* proj  = (__bf16*)take((size_t)M1 * TT * 2);   // 32 MB
    (void)ws_size;

    // 1) weight transposes/conversions to bf16 BT layout
    transpose_cvt_kernel<<<dim3(FF/32, HH/32, 1), 256, 0, stream>>>(
        pw, pwT, HH, FF, 0, 0);                       // pwT[f][h]
    transpose_cvt_kernel<<<dim3(HH/32, SS/32, 1), 256, 0, stream>>>(
        cw, cwT, SS, HH, 0, 0);                       // cwT[h][t]
    transpose_cvt_kernel<<<dim3(SS/32, SS/32, 1), 256, 0, stream>>>(
        sgu_w, sguT, SS, SS, 0, 0);                   // sguT[j][s]

    // 2) LN1 -> bf16
    ln1_kernel<<<dim3(M1), 256, 0, stream>>>(x, ng, nb, xln);

    // 3) GEMM1 + SiLU + split   (block tile 128x256)
    gemm1_silu_kernel<<<dim3(FF/256, M1/128), 256, 0, stream>>>(
        xln, pwT, pb, ubuf, vbuf);

    // 4) LN2 * gate-weight, in place on vbuf
    ln2_kernel<<<dim3(M1), 256, 0, stream>>>(vbuf, sng, snb, weight);

    // 5) per-batch transpose of m (f32) -> mT (bf16) for B operand
    transpose_cvt_kernel<<<dim3(TT/32, SS/32, BB), 256, 0, stream>>>(
        vbuf, mT, SS, TT, (long)SS * TT, (long)TT * SS);

    // 6) spatial GEMM + gate + u-multiply
    gemm2_gate_kernel<<<dim3(TT/256, SS/128, BB), 256, 0, stream>>>(
        sguT, mT, ubuf, weight, sgu_b, proj);

    // 7) output projection
    gemm3_out_kernel<<<dim3(HH/256, M1/128), 256, 0, stream>>>(
        proj, cwT, cb, out);
}
